// Qwen3MegaBlocksAdapter_16260746182725
// MI455X (gfx1250) — matrix operand order repaired and checked by reading, not compiled, not run on hardware
//
#include <hip/hip_runtime.h>

// ---------------- problem constants (from reference) ----------------
constexpr int Bsz = 1;
constexpr int S   = 1024;
constexpr int Dh  = 2048;   // hidden
constexpr int Eh  = 64;     // experts
constexpr int Kh  = 8;      // top-k
constexpr int Fh  = 768;    // intermediate
constexpr int Ch  = 256;    // capacity
constexpr int Th  = Bsz * S;

typedef __attribute__((ext_vector_type(16))) __bf16 v16bf;
typedef __attribute__((ext_vector_type(8)))  float  v8f;

// ---------------- CDNA5 async global->LDS copy (16B per lane) -------
// LDS address comes from vdst, global from vaddr; the instruction offset adds to both.
// Generic LDS addresses: low 32 bits are the LDS byte offset (aperture map).
__device__ inline void async_copy_b128x2(void* lds_dst, const void* gsrc) {
  const unsigned lds = (unsigned)(unsigned long long)(uintptr_t)lds_dst;
  const unsigned long long g = (unsigned long long)(uintptr_t)gsrc;
  asm volatile("global_load_async_to_lds_b128 %0, %1, off\n\t"
               "global_load_async_to_lds_b128 %0, %1, off offset:16"
               :: "v"(lds), "v"(g) : "memory");
}
__device__ inline void wait_async_lds() {
  asm volatile("s_wait_asynccnt 0" ::: "memory");
}

// ---------------- WMMA fragment helpers (wave32, 16x16x32 bf16) -----
// A-matrix 16x32 bf16: lane L holds row M=L%16; lanes 0-15 cover K 0..15,
// lanes 16-31 cover K 16..31 (contiguous, matching load_b_frag).
__device__ inline v16bf load_a_frag(const __bf16* As, int lane, int ld) {
  const int row  = lane & 15;
  const int koff = (lane >> 4) * 16;
  v16bf a;
#pragma unroll
  for (int i = 0; i < 8; ++i) {
    const int k = 2 * i + koff;
    a[2 * i]     = As[row * ld + k];
    a[2 * i + 1] = As[row * ld + k + 1];
  }
  return a;
}

// B-matrix 32x16 bf16 stored n-major in LDS (Bs[n][k]): lane n=L%16,
// lanes 0-15 hold K=0..15, lanes 16-31 hold K=16..31 -> one 32B vector load.
__device__ inline v16bf load_b_frag(const __bf16* Bs, int lane, int ld) {
  const int n     = lane & 15;
  const int kbase = (lane >> 4) * 16;
  return *reinterpret_cast<const v16bf*>(Bs + n * ld + kbase);
}

__device__ inline float fast_silu_mul(float g, float u) {
  // silu(g)*u with v_rcp_f32 instead of IEEE division refinement
  return g * __builtin_amdgcn_rcpf(1.0f + __expf(-g)) * u;
}

// ---------------- kernel 1: zero fill (vectorized) ------------------
__global__ void zero_kernel(uint4* __restrict__ p, size_t n16) {
  size_t i = (size_t)blockIdx.x * blockDim.x + threadIdx.x;
  const size_t stride = (size_t)gridDim.x * blockDim.x;
  const uint4 z = {0u, 0u, 0u, 0u};
  for (; i < n16; i += stride) p[i] = z;
}

// ---------------- kernel 2: router (softmax + top-8 + L1 norm) ------
__global__ __launch_bounds__(64) void router_kernel(
    const float* __restrict__ x, const float* __restrict__ rw,
    int* __restrict__ topk_idx, float* __restrict__ topk_wt) {
  __shared__ float logits[Eh];
  const int t = blockIdx.x;
  const int j = threadIdx.x;               // expert id
  const float* xr = x + (size_t)t * Dh;
  const float* wr = rw + (size_t)j * Dh;
  float acc = 0.0f;
  for (int d = 0; d < Dh; d += 4) {
    acc += xr[d] * wr[d] + xr[d + 1] * wr[d + 1] +
           xr[d + 2] * wr[d + 2] + xr[d + 3] * wr[d + 3];
  }
  logits[j] = acc;
  __syncthreads();
  if (j == 0) {
    float p[Eh];
    float mx = -3.0e38f;
    for (int i = 0; i < Eh; ++i) mx = fmaxf(mx, logits[i]);
    float sum = 0.0f;
    for (int i = 0; i < Eh; ++i) { p[i] = __expf(logits[i] - mx); sum += p[i]; }
    const float inv = 1.0f / sum;
    for (int i = 0; i < Eh; ++i) p[i] *= inv;
    int ids[Kh]; float wts[Kh]; float ssum = 0.0f;
    for (int k = 0; k < Kh; ++k) {
      int bi = 0; float bv = -1.0f;
      for (int i = 0; i < Eh; ++i)
        if (p[i] > bv) { bv = p[i]; bi = i; }
      ids[k] = bi; wts[k] = bv; ssum += bv; p[bi] = -2.0f;
    }
    const float is = 1.0f / ssum;
    for (int k = 0; k < Kh; ++k) {
      topk_idx[t * Kh + k] = ids[k];
      topk_wt[t * Kh + k]  = wts[k] * is;
    }
  }
}

// ---------------- kernel 3: stable slot assignment (wave32 ballot) --
__global__ __launch_bounds__(32) void pos_kernel(
    const int* __restrict__ topk_idx, int* __restrict__ pos) {
  const int e    = blockIdx.x;
  const int lane = threadIdx.x;
  int running = 0;
  for (int base = 0; base < Th * Kh; base += 32) {
    const int r = base + lane;
    const bool match = (topk_idx[r] == e);
    const unsigned mask = __builtin_amdgcn_ballot_w32(match);
    if (match) {
      const int p = running + __popc(mask & ((1u << lane) - 1u));
      pos[r] = (p < Ch) ? p : -1;   // capacity drop
    }
    running += __popc(mask);
  }
}

// ---------------- kernel 4: dispatch scatter (f32 -> bf16, 16B) -----
__global__ __launch_bounds__(256) void scatter_kernel(
    const float* __restrict__ x, const int* __restrict__ topk_idx,
    const int* __restrict__ pos, __bf16* __restrict__ xg) {
  const int r = blockIdx.x;
  const int p = pos[r];
  if (p < 0) return;
  const int e = topk_idx[r];
  const int t = r / Kh;
  const float* src = x + (size_t)t * Dh;
  __bf16* dst = xg + ((size_t)e * Ch + p) * Dh;
  const int d = threadIdx.x * 8;            // 256 threads * 8 = 2048
  const float4 f0 = *reinterpret_cast<const float4*>(src + d);
  const float4 f1 = *reinterpret_cast<const float4*>(src + d + 4);
  union { __bf16 h[8]; uint4 q; } u;
  u.h[0] = (__bf16)f0.x; u.h[1] = (__bf16)f0.y;
  u.h[2] = (__bf16)f0.z; u.h[3] = (__bf16)f0.w;
  u.h[4] = (__bf16)f1.x; u.h[5] = (__bf16)f1.y;
  u.h[6] = (__bf16)f1.z; u.h[7] = (__bf16)f1.w;
  *reinterpret_cast<uint4*>(dst + d) = u.q;
}

// ---------------- kernel 5: fused gate/up GEMM + SiLU-GLU -----------
// H[e,c,f] = silu(xg[e,c,:] . w1[e,f,:]) * (xg[e,c,:] . v1[e,f,:])
// 64x64 output tile per block, K-step 64 (2 WMMA k-halves per step).
__global__ __launch_bounds__(256) void gemm1_glu_kernel(
    const __bf16* __restrict__ xg, const float* __restrict__ w1,
    const float* __restrict__ v1, __bf16* __restrict__ Hout) {
  __shared__ __bf16 As [64 * 64];    // activations (async-copied, bf16)
  __shared__ __bf16 W1s[64 * 64];    // gate weights (f32->bf16 convert)
  __shared__ __bf16 V1s[64 * 64];    // up weights
  const int e  = blockIdx.z;
  const int c0 = blockIdx.y * 64;
  const int f0 = blockIdx.x * 64;
  const int tid  = threadIdx.x;
  const int wave = tid >> 5;
  const int lane = tid & 31;
  const int tm  = wave & 3;
  const int tn0 = (wave >> 2) * 2;

  v8f accG0 = {}; v8f accG1 = {}; v8f accU0 = {}; v8f accU1 = {};

  const int lrow = tid >> 2;            // 0..63
  const int lcol = (tid & 3) * 16;      // 0,16,32,48

  const __bf16* xgb = xg + ((size_t)e * Ch + c0) * Dh;
  const float*  w1b = w1 + ((size_t)e * Fh + f0) * Dh;
  const float*  v1b = v1 + ((size_t)e * Fh + f0) * Dh;

  for (int d0 = 0; d0 < Dh; d0 += 64) {
    // A tile 64x64 bf16: async DMA straight into LDS (2x16B per thread)
    async_copy_b128x2(&As[lrow * 64 + lcol],
                      xgb + (size_t)lrow * Dh + d0 + lcol);
    // weight tiles: fp32 -> bf16 convert on the fly (n-major in LDS)
#pragma unroll
    for (int jj = 0; jj < 16; ++jj) {
      W1s[lrow * 64 + lcol + jj] = (__bf16)w1b[(size_t)lrow * Dh + d0 + lcol + jj];
      V1s[lrow * 64 + lcol + jj] = (__bf16)v1b[(size_t)lrow * Dh + d0 + lcol + jj];
    }
    wait_async_lds();
    __syncthreads();

#pragma unroll
    for (int kk = 0; kk < 64; kk += 32) {
      const v16bf a   = load_a_frag(&As[tm * 16 * 64 + kk], lane, 64);
      const v16bf bg0 = load_b_frag(&W1s[(tn0 * 16) * 64 + kk], lane, 64);
      const v16bf bg1 = load_b_frag(&W1s[((tn0 + 1) * 16) * 64 + kk], lane, 64);
      const v16bf bu0 = load_b_frag(&V1s[(tn0 * 16) * 64 + kk], lane, 64);
      const v16bf bu1 = load_b_frag(&V1s[((tn0 + 1) * 16) * 64 + kk], lane, 64);
      accG0 = __builtin_amdgcn_wmma_f32_16x16x32_bf16(false, a, false, bg0, (short)0, accG0, false, false);
      accG1 = __builtin_amdgcn_wmma_f32_16x16x32_bf16(false, a, false, bg1, (short)0, accG1, false, false);
      accU0 = __builtin_amdgcn_wmma_f32_16x16x32_bf16(false, a, false, bu0, (short)0, accU0, false, false);
      accU1 = __builtin_amdgcn_wmma_f32_16x16x32_bf16(false, a, false, bu1, (short)0, accU1, false, false);
    }
    __syncthreads();
  }

  // epilogue: silu(G)*U, store bf16
  const int n     = lane & 15;
  const int mbase = (lane >> 4) * 8;
#pragma unroll
  for (int v = 0; v < 8; ++v) {
    const int crow = c0 + tm * 16 + mbase + v;
    Hout[((size_t)e * Ch + crow) * Fh + f0 + tn0 * 16 + n] =
        (__bf16)fast_silu_mul(accG0[v], accU0[v]);
    Hout[((size_t)e * Ch + crow) * Fh + f0 + (tn0 + 1) * 16 + n] =
        (__bf16)fast_silu_mul(accG1[v], accU1[v]);
  }
}

// ---------------- kernel 6: down-proj GEMM  y = H @ w2 --------------
// y[e,c,d] = sum_f H[e,c,f] * w2[e,f,d] ; K-step 64 over F.
__global__ __launch_bounds__(256) void gemm2_down_kernel(
    const __bf16* __restrict__ Hbuf, const float* __restrict__ w2,
    __bf16* __restrict__ ybuf) {
  __shared__ __bf16 Hs [64 * 64];
  __shared__ __bf16 W2s[64 * 64];   // [n=64][k=64] (n-major)
  const int e  = blockIdx.z;
  const int c0 = blockIdx.y * 64;
  const int d0 = blockIdx.x * 64;
  const int tid  = threadIdx.x;
  const int wave = tid >> 5;
  const int lane = tid & 31;
  const int tm  = wave & 3;
  const int tn0 = (wave >> 2) * 2;

  v8f acc0 = {}; v8f acc1 = {};

  const int lrow = tid >> 2;            // 0..63
  const int lcol = (tid & 3) * 16;      // 0,16,32,48

  const __bf16* Hb = Hbuf + ((size_t)e * Ch + c0) * Fh;

  for (int f0 = 0; f0 < Fh; f0 += 64) {
    // A tile 64x64 bf16: async DMA into LDS
    async_copy_b128x2(&Hs[lrow * 64 + lcol],
                      Hb + (size_t)lrow * Fh + f0 + lcol);
    // B tile: W2s[n][k] = w2[e, f0+k, d0+n]; coalesced reads along d
    {
      const int k  = tid >> 2;          // 0..63
      const int n0 = (tid & 3) * 16;    // 0..48
      const float* src = w2 + ((size_t)e * Fh + f0 + k) * Dh + d0 + n0;
#pragma unroll
      for (int jj = 0; jj < 16; ++jj)
        W2s[(n0 + jj) * 64 + k] = (__bf16)src[jj];
    }
    wait_async_lds();
    __syncthreads();

#pragma unroll
    for (int kk = 0; kk < 64; kk += 32) {
      const v16bf a  = load_a_frag(&Hs[tm * 16 * 64 + kk], lane, 64);
      const v16bf b0 = load_b_frag(&W2s[(tn0 * 16) * 64 + kk], lane, 64);
      const v16bf b1 = load_b_frag(&W2s[((tn0 + 1) * 16) * 64 + kk], lane, 64);
      acc0 = __builtin_amdgcn_wmma_f32_16x16x32_bf16(false, a, false, b0, (short)0, acc0, false, false);
      acc1 = __builtin_amdgcn_wmma_f32_16x16x32_bf16(false, a, false, b1, (short)0, acc1, false, false);
    }
    __syncthreads();
  }

  const int n     = lane & 15;
  const int mbase = (lane >> 4) * 8;
#pragma unroll
  for (int v = 0; v < 8; ++v) {
    const int crow = c0 + tm * 16 + mbase + v;
    ybuf[((size_t)e * Ch + crow) * Dh + d0 + tn0 * 16 + n]       = (__bf16)acc0[v];
    ybuf[((size_t)e * Ch + crow) * Dh + d0 + (tn0 + 1) * 16 + n] = (__bf16)acc1[v];
  }
}

// ---------------- kernel 7: weighted combine (deterministic) --------
__global__ __launch_bounds__(256) void combine_kernel(
    const __bf16* __restrict__ ybuf, const int* __restrict__ topk_idx,
    const int* __restrict__ pos, const float* __restrict__ topk_wt,
    float* __restrict__ out) {
  const int t = blockIdx.x;
  const int d = threadIdx.x * 8;            // contiguous 16B per thread
  float acc[8];
#pragma unroll
  for (int i = 0; i < 8; ++i) acc[i] = 0.0f;
  for (int k = 0; k < Kh; ++k) {
    const int r = t * Kh + k;
    const int p = pos[r];
    if (p < 0) continue;
    const int e = topk_idx[r];
    const float w = topk_wt[r];
    const uint4 q = *reinterpret_cast<const uint4*>(
        ybuf + ((size_t)e * Ch + p) * Dh + d);
    const __bf16* yb = reinterpret_cast<const __bf16*>(&q);
#pragma unroll
    for (int i = 0; i < 8; ++i) acc[i] += w * (float)yb[i];
  }
  float4 o0 = {acc[0], acc[1], acc[2], acc[3]};
  float4 o1 = {acc[4], acc[5], acc[6], acc[7]};
  *reinterpret_cast<float4*>(out + (size_t)t * Dh + d)     = o0;
  *reinterpret_cast<float4*>(out + (size_t)t * Dh + d + 4) = o1;
}

// ---------------- host: launch pipeline -----------------------------
extern "C" void kernel_launch(void* const* d_in, const int* in_sizes, int n_in,
                              void* d_out, int out_size, void* d_ws, size_t ws_size,
                              hipStream_t stream) {
  (void)in_sizes; (void)n_in; (void)out_size; (void)ws_size;
  const float* x   = (const float*)d_in[0];   // [1,1024,2048]
  const float* rw  = (const float*)d_in[1];   // [64,2048]
  const float* w1  = (const float*)d_in[2];   // [64,768,2048]
  const float* v1  = (const float*)d_in[3];   // [64,768,2048]
  const float* w2  = (const float*)d_in[4];   // [64,768,2048]
  float* out = (float*)d_out;                 // [1,1024,2048]

  char* ws = (char*)d_ws;
  size_t off = 0;
  auto suballoc = [&](size_t bytes) -> void* {
    void* p = ws + off;
    off = (off + bytes + 255) & ~(size_t)255;
    return p;
  };
  __bf16* xg   = (__bf16*)suballoc((size_t)Eh * Ch * Dh * sizeof(__bf16)); // 67 MB
  __bf16* Hbuf = (__bf16*)suballoc((size_t)Eh * Ch * Fh * sizeof(__bf16)); // 25 MB
  __bf16* ybuf = (__bf16*)suballoc((size_t)Eh * Ch * Dh * sizeof(__bf16)); // 67 MB
  int*    tidx = (int*)  suballoc((size_t)Th * Kh * sizeof(int));
  float*  twt  = (float*)suballoc((size_t)Th * Kh * sizeof(float));
  int*    tpos = (int*)  suballoc((size_t)Th * Kh * sizeof(int));

  // 1) zero the dispatch buffer (unused capacity slots must be 0)
  zero_kernel<<<4096, 256, 0, stream>>>(
      (uint4*)xg, (size_t)Eh * Ch * Dh * sizeof(__bf16) / 16);
  // 2) router: logits -> softmax -> top-8 -> L1 normalize
  router_kernel<<<Th, 64, 0, stream>>>(x, rw, tidx, twt);
  // 3) stable per-expert slot assignment
  pos_kernel<<<Eh, 32, 0, stream>>>(tidx, tpos);
  // 4) dispatch tokens into [E,C,D] bf16
  scatter_kernel<<<Th * Kh, 256, 0, stream>>>(x, tidx, tpos, xg);
  // 5) fused gate/up GEMM + GLU  (WMMA bf16, async LDS staging)
  gemm1_glu_kernel<<<dim3(Fh / 64, Ch / 64, Eh), 256, 0, stream>>>(xg, w1, v1, Hbuf);
  // 6) down projection (WMMA bf16, async LDS staging)
  gemm2_down_kernel<<<dim3(Dh / 64, Ch / 64, Eh), 256, 0, stream>>>(Hbuf, w2, ybuf);
  // 7) weighted combine back to tokens
  combine_kernel<<<Th, 256, 0, stream>>>(ybuf, tidx, tpos, twt, out);
}
